// CustomLSTM_22883585753321
// MI455X (gfx1250) — matrix operand order repaired and checked by reading, not compiled, not run on hardware
//
#include <hip/hip_runtime.h>

// ---------- types ----------
typedef __attribute__((ext_vector_type(16))) __bf16 bf16x16;
typedef __attribute__((ext_vector_type(8)))  float  f32x8;
typedef __attribute__((ext_vector_type(4)))  unsigned int u32x4;
typedef __attribute__((ext_vector_type(8)))  int i32x8;
typedef __attribute__((ext_vector_type(4)))  int i32x4;

#if __has_builtin(__builtin_amdgcn_tensor_load_to_lds)
#define HAVE_TDM 1
#else
#define HAVE_TDM 0
#endif

union Frag { unsigned int u[8]; bf16x16 v; };

__device__ __forceinline__ unsigned short f2bf(float f) {
    unsigned int u = __builtin_bit_cast(unsigned int, f);
    unsigned int r = u + 0x7FFFu + ((u >> 16) & 1u);   // RNE
    return (unsigned short)(r >> 16);
}
__device__ __forceinline__ float sigm(float x) { return 1.0f / (1.0f + __expf(-x)); }

// Problem sizes
#define BB   64
#define SS   512
#define HH   512
#define G4H  2048
#define MTOT (BB * SS)

#if HAVE_TDM
__device__ __forceinline__ unsigned lds_off(const void* p) {
    return (unsigned)(unsigned long long)(uintptr_t)p;   // low 32 bits = LDS byte offset
}
// D# group1: data_size=2B, pad 4 DWORDs after every 16 DWORDs (row 64B -> stride 80B)
__device__ __forceinline__ i32x8 tdm_g1(unsigned dim0, unsigned dim1,
                                        unsigned t0, unsigned t1, unsigned t2,
                                        unsigned long long s0, unsigned long long s1)
{
    i32x8 g;
    g[0] = (int)((1u << 16) | (1u << 20) | (3u << 22) | (3u << 25));
    g[1] = (int)((dim0 & 0xFFFFu) << 16);
    g[2] = (int)((dim0 >> 16) | ((dim1 & 0xFFFFu) << 16));
    g[3] = (int)((dim1 >> 16) | (t0 << 16));
    g[4] = (int)(t1 | (t2 << 16));
    g[5] = (int)(s0 & 0xFFFFFFFFull);
    g[6] = (int)((unsigned)(s0 >> 32) | (((unsigned)(s1 & 0xFFFFull)) << 16));
    g[7] = (int)(s1 >> 16);
    return g;
}
__device__ __forceinline__ void tdm_load(unsigned lds, unsigned long long ga,
                                         i32x8 g1, i32x4 g2)
{
    u32x4 g0;
    g0[0] = 1u;                                             // count=1, user mode
    g0[1] = lds;                                            // lds_addr
    g0[2] = (unsigned)(ga & 0xFFFFFFFFull);                 // global_addr lo
    g0[3] = ((unsigned)((ga >> 32) & 0x01FFFFFFull)) | 0x80000000u;  // addr hi | type=2
    i32x4 g3 = {0, 0, 0, 0};
#if defined(__clang_major__) && __clang_major__ >= 23
    i32x8 z8 = {0, 0, 0, 0, 0, 0, 0, 0};
    __builtin_amdgcn_tensor_load_to_lds(g0, g1, g2, g3, z8, 0);
#else
    __builtin_amdgcn_tensor_load_to_lds(g0, g1, g2, g3, 0);
#endif
}
#endif // HAVE_TDM

// =====================================================================
// Kernel 0: prep — convert x, W_ih, W_hh to bf16; h0 -> bf16; copy c0
// =====================================================================
__global__ __launch_bounds__(256)
void prep_kernel(const float* __restrict__ x,  const float* __restrict__ Wih,
                 const float* __restrict__ Whh, const float* __restrict__ h0,
                 const float* __restrict__ c0,
                 unsigned short* __restrict__ xbf, unsigned short* __restrict__ wihbf,
                 unsigned short* __restrict__ whhbf, unsigned short* __restrict__ hbf,
                 float* __restrict__ c_buf)
{
    int idx = blockIdx.x * 256 + threadIdx.x;
    if (idx < MTOT * HH)  xbf[idx] = f2bf(x[idx]);
    if (idx < G4H * HH) { wihbf[idx] = f2bf(Wih[idx]); whhbf[idx] = f2bf(Whh[idx]); }
    if (idx < BB * HH)  { hbf[idx] = f2bf(h0[idx]); c_buf[idx] = c0[idx]; }
}

// =====================================================================
// Kernel 1: wx = x @ W_ih^T + b_ih -> [32768, 2048] f32
// 128x128 tile, K-chunks of 32, TDM double-buffered into padded LDS.
// =====================================================================
__global__ __launch_bounds__(256)
void wx_gemm(const unsigned short* __restrict__ xbf,
             const unsigned short* __restrict__ wihbf,
             const float* __restrict__ bih, float* __restrict__ wx)
{
    __shared__ unsigned short ldsA[2][128 * 40];
    __shared__ unsigned short ldsB[2][128 * 40];

    const int tid = threadIdx.x;
    const int lane = tid & 31, wave = tid >> 5;
    const int m0 = blockIdx.x * 128, n0 = blockIdx.y * 128;
    const int wm = wave >> 2, wn = wave & 3;
    const int hf = lane >> 4, l15 = lane & 15;

    f32x8 acc[4][2] = {};

#if HAVE_TDM
    const unsigned long long gA0 = (unsigned long long)(uintptr_t)xbf   + (unsigned long long)m0 * HH * 2ull;
    const unsigned long long gB0 = (unsigned long long)(uintptr_t)wihbf + (unsigned long long)n0 * HH * 2ull;
    const i32x8 g1a = tdm_g1(HH, MTOT, 32, 128, 0, HH, 0);
    const i32x8 g1b = tdm_g1(HH, G4H,  32, 128, 0, HH, 0);
    const i32x4 g2z = {0, 0, 0, 0};
    if (wave == 0) {
        tdm_load(lds_off(&ldsA[0][0]), gA0, g1a, g2z);
        tdm_load(lds_off(&ldsB[0][0]), gB0, g1b, g2z);
    }
#endif

    for (int c = 0; c < 16; ++c) {
        const int kc = c * 32;
#if HAVE_TDM
        const int b = c & 1;
        __syncthreads();                       // readers of buf[b^1] (iter c-1) done
        if (wave == 0) {
            if (c < 15) {
                tdm_load(lds_off(&ldsA[b ^ 1][0]), gA0 + (unsigned long long)(kc + 32) * 2, g1a, g2z);
                tdm_load(lds_off(&ldsB[b ^ 1][0]), gB0 + (unsigned long long)(kc + 32) * 2, g1b, g2z);
                __builtin_amdgcn_s_wait_tensorcnt(2);   // chunk c complete (in-order)
            } else {
                __builtin_amdgcn_s_wait_tensorcnt(0);
            }
        }
        __syncthreads();
        const unsigned short* bufA = ldsA[b];
        const unsigned short* bufB = ldsB[b];
#else
        const int lr = tid >> 1, lc = (tid & 1) * 16;
        uint4 a0 = *(const uint4*)(xbf   + (size_t)(m0 + lr) * HH + kc + lc);
        uint4 a1 = *(const uint4*)(xbf   + (size_t)(m0 + lr) * HH + kc + lc + 8);
        uint4 b0 = *(const uint4*)(wihbf + (size_t)(n0 + lr) * HH + kc + lc);
        uint4 b1 = *(const uint4*)(wihbf + (size_t)(n0 + lr) * HH + kc + lc + 8);
        __syncthreads();
        *(uint4*)&ldsA[0][lr * 40 + lc]     = a0;
        *(uint4*)&ldsA[0][lr * 40 + lc + 8] = a1;
        *(uint4*)&ldsB[0][lr * 40 + lc]     = b0;
        *(uint4*)&ldsB[0][lr * 40 + lc + 8] = b1;
        __syncthreads();
        const unsigned short* bufA = ldsA[0];
        const unsigned short* bufB = ldsB[0];
#endif
        Frag a[4], bfr[2];
        #pragma unroll
        for (int mi = 0; mi < 4; mi++) {
            const int m = wm * 64 + mi * 16 + l15;
            #pragma unroll
            for (int v = 0; v < 8; v++) {
                const int k0 = ((v & 3) << 1) + (hf << 3) + ((v >> 2) << 4);
                a[mi].u[v] = *(const unsigned int*)&bufA[m * 40 + k0];
            }
        }
        #pragma unroll
        for (int nj = 0; nj < 2; nj++) {
            const int n = wn * 32 + nj * 16 + l15;
            #pragma unroll
            for (int v = 0; v < 8; v++) {
                const int k0 = ((v & 3) << 1) + (hf << 3) + ((v >> 2) << 4);
                bfr[nj].u[v] = *(const unsigned int*)&bufB[n * 40 + k0];
            }
        }
        #pragma unroll
        for (int mi = 0; mi < 4; mi++)
            #pragma unroll
            for (int nj = 0; nj < 2; nj++)
                acc[mi][nj] = __builtin_amdgcn_wmma_f32_16x16x32_bf16(
                    false, a[mi].v, false, bfr[nj].v, (short)0, acc[mi][nj], false, false);
    }

    #pragma unroll
    for (int mi = 0; mi < 4; mi++)
        #pragma unroll
        for (int nj = 0; nj < 2; nj++) {
            const int n = n0 + wn * 32 + nj * 16 + l15;
            const float bias = bih[n];
            #pragma unroll
            for (int r = 0; r < 8; r++) {
                const int m = m0 + wm * 64 + mi * 16 + r + hf * 8;
                wx[(size_t)m * G4H + n] = acc[mi][nj][r] + bias;
            }
        }
}

// =====================================================================
// Kernel 2: one LSTM step. 8 WGs x 64 cols; wave owns 16 rows x 32 cols
// x all 4 gates -> nonlinearity entirely on WMMA accumulators.
// W tile fetched as a single 3-D TDM op (32 K x 64 rows x 4 gates).
// =====================================================================
__global__ __launch_bounds__(256)
void lstm_step(const float* __restrict__ wx, const float* __restrict__ bhh,
               const unsigned short* __restrict__ whh_bf,
               const unsigned short* __restrict__ h_in,
               unsigned short* __restrict__ h_out,
               float* __restrict__ c_buf, float* __restrict__ out, int step)
{
    __shared__ unsigned short ldsH[2][64  * 40];
    __shared__ unsigned short ldsW[2][256 * 40];

    const int tid = threadIdx.x;
    const int lane = tid & 31, wave = tid >> 5;
    const int j0 = blockIdx.x * 64;
    const int mi = wave & 3, nh = wave >> 2;
    const int hf = lane >> 4, l15 = lane & 15;

    f32x8 acc[4][2] = {};

#if HAVE_TDM
    const unsigned long long gH0 = (unsigned long long)(uintptr_t)h_in;
    const unsigned long long gW0 = (unsigned long long)(uintptr_t)whh_bf + (unsigned long long)j0 * HH * 2ull;
    const i32x8 g1h = tdm_g1(HH, BB,  32, 64, 0, HH, 0);
    const i32x8 g1w = tdm_g1(HH, G4H, 32, 64, 4, HH, (unsigned long long)HH * HH);
    const i32x4 g2z = {0, 0, 0, 0};
    const i32x4 g2w = {4, 0, 0, 0};          // tensor_dim2 = 4 gates
    if (wave == 0) {
        tdm_load(lds_off(&ldsH[0][0]), gH0, g1h, g2z);
        tdm_load(lds_off(&ldsW[0][0]), gW0, g1w, g2w);
    }
#endif

    for (int c = 0; c < 16; ++c) {
        const int kc = c * 32;
#if HAVE_TDM
        const int b = c & 1;
        __syncthreads();
        if (wave == 0) {
            if (c < 15) {
                tdm_load(lds_off(&ldsH[b ^ 1][0]), gH0 + (unsigned long long)(kc + 32) * 2, g1h, g2z);
                tdm_load(lds_off(&ldsW[b ^ 1][0]), gW0 + (unsigned long long)(kc + 32) * 2, g1w, g2w);
                __builtin_amdgcn_s_wait_tensorcnt(2);
            } else {
                __builtin_amdgcn_s_wait_tensorcnt(0);
            }
        }
        __syncthreads();
        const unsigned short* bufH = ldsH[b];
        const unsigned short* bufW = ldsW[b];
#else
        const int hr = tid >> 2, hc = (tid & 3) * 8;
        const int wg_ = tid >> 6, wlr = tid & 63;
        uint4 hv  = *(const uint4*)(h_in + hr * HH + kc + hc);
        uint4 wv0 = *(const uint4*)(whh_bf + (size_t)(wg_ * HH + j0 + wlr) * HH + kc);
        uint4 wv1 = *(const uint4*)(whh_bf + (size_t)(wg_ * HH + j0 + wlr) * HH + kc + 8);
        __syncthreads();
        *(uint4*)&ldsH[0][hr * 40 + hc]              = hv;
        *(uint4*)&ldsW[0][(wg_ * 64 + wlr) * 40]     = wv0;
        *(uint4*)&ldsW[0][(wg_ * 64 + wlr) * 40 + 8] = wv1;
        __syncthreads();
        const unsigned short* bufH = ldsH[0];
        const unsigned short* bufW = ldsW[0];
#endif
        Frag a, bf2[4][2];
        {
            const int m = mi * 16 + l15;
            #pragma unroll
            for (int v = 0; v < 8; v++) {
                const int k0 = ((v & 3) << 1) + (hf << 3) + ((v >> 2) << 4);
                a.u[v] = *(const unsigned int*)&bufH[m * 40 + k0];
            }
        }
        #pragma unroll
        for (int g = 0; g < 4; g++)
            #pragma unroll
            for (int nj = 0; nj < 2; nj++) {
                const int rr = g * 64 + nh * 32 + nj * 16 + l15;
                #pragma unroll
                for (int v = 0; v < 8; v++) {
                    const int k0 = ((v & 3) << 1) + (hf << 3) + ((v >> 2) << 4);
                    bf2[g][nj].u[v] = *(const unsigned int*)&bufW[rr * 40 + k0];
                }
            }
        #pragma unroll
        for (int g = 0; g < 4; g++)
            #pragma unroll
            for (int nj = 0; nj < 2; nj++)
                acc[g][nj] = __builtin_amdgcn_wmma_f32_16x16x32_bf16(
                    false, a.v, false, bf2[g][nj].v, (short)0, acc[g][nj], false, false);
    }

    #pragma unroll
    for (int nj = 0; nj < 2; nj++) {
        const int jl = nh * 32 + nj * 16 + l15;
        const int j  = j0 + jl;
        const float bi  = bhh[j];
        const float bff = bhh[HH + j];
        const float bg  = bhh[2 * HH + j];
        const float bo  = bhh[3 * HH + j];
        #pragma unroll
        for (int r = 0; r < 8; r++) {
            const int m = mi * 16 + r + hf * 8;
            const size_t wxrow = ((size_t)m * SS + step) * G4H;
            float iv = acc[0][nj][r] + wx[wxrow + j]          + bi;
            float fv = acc[1][nj][r] + wx[wxrow + HH + j]     + bff;
            float gv = acc[2][nj][r] + wx[wxrow + 2 * HH + j] + bg;
            float ov = acc[3][nj][r] + wx[wxrow + 3 * HH + j] + bo;
            float c_old = c_buf[m * HH + j];
            float c_new = sigm(fv) * c_old + sigm(iv) * tanhf(gv);
            float hv    = sigm(ov) * tanhf(c_new);
            c_buf[m * HH + j] = c_new;
            out[((size_t)m * SS + step) * HH + j] = hv;
            h_out[m * HH + j] = f2bf(hv);
        }
    }
}

// =====================================================================
// Kernel 3: finalize — copy final h (== out[:,511,:]) and c into tail
// =====================================================================
__global__ __launch_bounds__(256)
void fin_kernel(float* __restrict__ out, const float* __restrict__ c_buf)
{
    int idx = blockIdx.x * 256 + threadIdx.x;
    if (idx < BB * HH) {
        int b = idx >> 9, j = idx & 511;
        out[(size_t)BB * SS * HH + idx]           = out[((size_t)b * SS + (SS - 1)) * HH + j];
        out[(size_t)BB * SS * HH + BB * HH + idx] = c_buf[idx];
    }
}

// =====================================================================
extern "C" void kernel_launch(void* const* d_in, const int* in_sizes, int n_in,
                              void* d_out, int out_size, void* d_ws, size_t ws_size,
                              hipStream_t stream)
{
    const float* x   = (const float*)d_in[0];
    const float* h0  = (const float*)d_in[1];
    const float* c0  = (const float*)d_in[2];
    const float* Wih = (const float*)d_in[3];
    const float* bih = (const float*)d_in[4];
    const float* Whh = (const float*)d_in[5];
    const float* bhh = (const float*)d_in[6];
    float* out = (float*)d_out;

    // workspace layout
    char* wsp = (char*)d_ws;
    float*          wx     = (float*)wsp;                                   // 256 MB
    unsigned short* xbf    = (unsigned short*)(wsp + (size_t)MTOT * G4H * 4);        // 32 MB
    unsigned short* wihbf  = xbf   + (size_t)MTOT * HH;                     // 2 MB
    unsigned short* whhbf  = wihbf + (size_t)G4H * HH;                      // 2 MB
    unsigned short* h0b    = whhbf + (size_t)G4H * HH;                      // 64 KB
    unsigned short* h1b    = h0b   + (size_t)BB * HH;                       // 64 KB
    float*          c_buf  = (float*)(h1b + (size_t)BB * HH);               // 128 KB

    prep_kernel<<<(MTOT * HH) / 256, 256, 0, stream>>>(x, Wih, Whh, h0, c0,
                                                       xbf, wihbf, whhbf, h0b, c_buf);

    wx_gemm<<<dim3(MTOT / 128, G4H / 128), 256, 0, stream>>>(xbf, wihbf, bih, wx);

    for (int t = 0; t < SS; t++) {
        const unsigned short* hin = (t & 1) ? h1b : h0b;
        unsigned short*      hout = (t & 1) ? h0b : h1b;
        lstm_step<<<HH / 64, 256, 0, stream>>>(wx, bhh, whhbf, hin, hout, c_buf, out, t);
    }

    fin_kernel<<<(BB * HH + 255) / 256, 256, 0, stream>>>(out, c_buf);
}
